// PGN_936302871148
// MI455X (gfx1250) — matrix operand order repaired and checked by reading, not compiled, not run on hardware
//
#include <hip/hip_runtime.h>
#include <math.h>

// ---------------------------------------------------------------------------
// Pointer-generator decoder for MI455X (gfx1250, wave32, WMMA + TDM).
// - All GEMMs: v_wmma_f32_16x16x32_f16, f32 accumulate; weights converted to
//   f16 transposed (N-major) once per call so fragments load as 16B chunks.
// - Dominant GEMM (h@Wout) and enc_feat GEMM stage B strips through LDS with
//   TENSOR_LOAD_TO_LDS (Tensor Data Mover), double-buffered, two waves per
//   block sharing the staged strip; sync via s_wait_tensorcnt + barriers.
// ---------------------------------------------------------------------------

typedef __attribute__((ext_vector_type(16))) _Float16     v16h;
typedef __attribute__((ext_vector_type(8)))  float        v8f;
typedef __attribute__((ext_vector_type(4)))  float        f32x4;
typedef __attribute__((ext_vector_type(4)))  unsigned int u32x4;
typedef __attribute__((ext_vector_type(8)))  int          i32x8;
typedef __attribute__((ext_vector_type(4)))  int          i32x4;

namespace {

constexpr int Bsz = 32, Lsz = 400, Tsteps = 40, Hd = 512, Ad = 512, Ed = 256,
              Vd = 32000, OOV = 50, Dd = Ed + Hd, Vext = Vd + OOV;

// ---------------- reductions (wave32) ----------------
__device__ __forceinline__ float wredSum(float v) {
#pragma unroll
  for (int o = 16; o; o >>= 1) v += __shfl_xor(v, o, 32);
  return v;
}
__device__ __forceinline__ float wredMax(float v) {
#pragma unroll
  for (int o = 16; o; o >>= 1) v = fmaxf(v, __shfl_xor(v, o, 32));
  return v;
}
__device__ __forceinline__ float bredSum(float v, float* sh, int nw) {
  const int lane = threadIdx.x & 31, w = threadIdx.x >> 5;
  v = wredSum(v);
  if (lane == 0) sh[w] = v;
  __syncthreads();
  float r = ((int)threadIdx.x < nw) ? sh[threadIdx.x] : 0.f;
  if (w == 0) { r = wredSum(r); if (lane == 0) sh[0] = r; }
  __syncthreads();
  r = sh[0];
  __syncthreads();
  return r;
}
__device__ __forceinline__ float bredMax(float v, float* sh, int nw) {
  const int lane = threadIdx.x & 31, w = threadIdx.x >> 5;
  v = wredMax(v);
  if (lane == 0) sh[w] = v;
  __syncthreads();
  float r = ((int)threadIdx.x < nw) ? sh[threadIdx.x] : -INFINITY;
  if (w == 0) { r = wredMax(r); if (lane == 0) sh[0] = r; }
  __syncthreads();
  r = sh[0];
  __syncthreads();
  return r;
}

// ---------------- conversion helpers ----------------
__global__ void k_f32_to_f16(const float* __restrict__ X,
                             _Float16* __restrict__ Y, size_t n) {
  size_t i = (size_t)blockIdx.x * blockDim.x + threadIdx.x;
  if (i < n) Y[i] = (_Float16)X[i];
}

// W: K x N (row-major) -> Wt: N x K (row-major), f16
__global__ void k_f32_to_f16_T(const float* __restrict__ W,
                               _Float16* __restrict__ Wt, int K, int N) {
  size_t i = (size_t)blockIdx.x * blockDim.x + threadIdx.x;
  if (i >= (size_t)K * N) return;
  int k = (int)(i / N), n = (int)(i % N);
  Wt[(size_t)n * K + k] = (_Float16)W[i];
}

__global__ void k_copy_f32(const float* __restrict__ s, float* __restrict__ d,
                           size_t n) {
  size_t i = (size_t)blockIdx.x * blockDim.x + threadIdx.x;
  if (i < n) d[i] = s[i];
}
__global__ void k_zero_f32(float* __restrict__ d, size_t n) {
  size_t i = (size_t)blockIdx.x * blockDim.x + threadIdx.x;
  if (i < n) d[i] = 0.f;
}

// ---------------- direct-register WMMA GEMM ----------------
// C(MxN) = A(MxK) * Bt(NxK)^T + bias.  One wave -> 16x64 strip (A reuse x4).
__global__ __launch_bounds__(32) void wmma_gemm_nt(
    const _Float16* __restrict__ A, const _Float16* __restrict__ Bt,
    const float* __restrict__ bias, float* __restrict__ C,
    int M, int N, int K) {
  const int lane = threadIdx.x;
  const int g  = lane >> 4;
  const int rr = lane & 15;
  const int m0 = blockIdx.y << 4;
  const int n0 = blockIdx.x << 6;
  (void)M;

  const _Float16* arow = A + (size_t)(m0 + rr) * K;
  const _Float16* brow[4];
  v8f acc[4] = {};
#pragma unroll
  for (int nt = 0; nt < 4; ++nt)
    brow[nt] = Bt + (size_t)(n0 + nt * 16 + rr) * K;

  for (int k0 = 0; k0 < K; k0 += 32) {
    v16h a, b[4];
    f32x4* ap = (f32x4*)&a;
    ap[0] = *(const f32x4*)(arow + k0 + 16 * g);
    ap[1] = *(const f32x4*)(arow + k0 + 16 * g + 8);
    const int bo = k0 + 16 * g;
#pragma unroll
    for (int nt = 0; nt < 4; ++nt) {     // load all fragments first...
      f32x4* bpv = (f32x4*)&b[nt];
      bpv[0] = *(const f32x4*)(brow[nt] + bo);
      bpv[1] = *(const f32x4*)(brow[nt] + bo + 8);
    }
#pragma unroll
    for (int nt = 0; nt < 4; ++nt)       // ...then issue WMMAs (partial waits)
      acc[nt] = __builtin_amdgcn_wmma_f32_16x16x32_f16(
          false, a, false, b[nt], (short)0, acc[nt], false, false);
  }

#pragma unroll
  for (int nt = 0; nt < 4; ++nt) {
    const int n = n0 + nt * 16 + rr;
    const float bb = bias ? bias[n] : 0.f;
#pragma unroll
    for (int e = 0; e < 8; ++e)
      C[(size_t)(m0 + e + 8 * g) * N + n] = acc[nt][e] + bb;
  }
}

// ---------------- TDM helper: DMA a 64-row x 32-half B chunk into LDS ------
// D# per CDNA5 ISA; 2D tile, data_size=2B,
// tile_dim0=32 (K chunk), tile_dim1=64 (N rows), dim0 stride = K.
// This toolchain declares the 6-arg builtin (extra int32x8 group, then cpol).
__device__ __forceinline__ void tdm_load_chunk(const _Float16* gsrc, int K,
                                               int k_rem, unsigned lds_addr) {
  const unsigned long long ga = (unsigned long long)(uintptr_t)gsrc;
  u32x4 g0;
  g0[0] = 1u;                                   // count=1, user descriptor
  g0[1] = lds_addr;                             // lds_addr (bytes)
  g0[2] = (unsigned)ga;                         // global_addr[31:0]
  g0[3] = (unsigned)(ga >> 32) | (2u << 30);    // global_addr[56:32] | type=2
  const unsigned dim0 = (unsigned)k_rem;        // elements left along K
  const unsigned dim1 = 64u;                    // rows
  i32x8 g1;
  g1[0] = (int)(1u << 16);                      // data_size=1 (2 bytes/elem)
  g1[1] = (int)((dim0 & 0xFFFFu) << 16);        // tensor_dim0[15:0]
  g1[2] = (int)((dim0 >> 16) | ((dim1 & 0xFFFFu) << 16));   // dim0 hi | dim1 lo
  g1[3] = (int)((dim1 >> 16) | (32u << 16));    // dim1 hi | tile_dim0=32
  g1[4] = (int)64u;                             // tile_dim1=64, tile_dim2=0
  g1[5] = (int)(unsigned)K;                     // tensor_dim0_stride[31:0]
  g1[6] = 0;                                    // stride hi | dim1_stride lo
  g1[7] = 0;
  const i32x4 z4 = {0, 0, 0, 0};
  const i32x8 z8 = {0, 0, 0, 0, 0, 0, 0, 0};
  __builtin_amdgcn_tensor_load_to_lds(g0, g1, z4, z4, z8, 0);
}

// ---------------- LDS-staged WMMA GEMM (TDM double buffer) ----------------
// Block = 64 threads (2 waves). Wave w owns M-tile (blockIdx.y*2 + w); both
// waves consume the same B strip staged in LDS by the Tensor Data Mover.
__global__ __launch_bounds__(64) void wmma_gemm_tdm(
    const _Float16* __restrict__ A, const _Float16* __restrict__ Bt,
    const float* __restrict__ bias, float* __restrict__ C,
    int M, int N, int K) {
  __shared__ __align__(16) _Float16 lds[2][64 * 32];
  const int wave = threadIdx.x >> 5;
  const int lane = threadIdx.x & 31;
  const int g  = lane >> 4;
  const int rr = lane & 15;
  const int m0 = (blockIdx.y * 2 + wave) << 4;
  const int n0 = blockIdx.x << 6;
  (void)M;

  const _Float16* arow  = A + (size_t)(m0 + rr) * K;
  const _Float16* strip = Bt + (size_t)n0 * K;   // 64 rows x K halves
  const unsigned lds_a[2] = {(unsigned)(uintptr_t)&lds[0][0],
                             (unsigned)(uintptr_t)&lds[1][0]};
  v8f acc[4] = {};

  const int nchunks = K >> 5;
  if (wave == 0) tdm_load_chunk(strip, K, K, lds_a[0]);

  for (int i = 0; i < nchunks; ++i) {
    const int k0 = i << 5;
    if (wave == 0) {
      if (i + 1 < nchunks) {
        tdm_load_chunk(strip + (k0 + 32), K, K - (k0 + 32), lds_a[(i + 1) & 1]);
        __builtin_amdgcn_s_wait_tensorcnt(1);   // chunk i landed
      } else {
        __builtin_amdgcn_s_wait_tensorcnt(0);
      }
    }
    __syncthreads();                            // buf[i&1] visible to both waves

    const _Float16* lbuf = &lds[i & 1][0];
    v16h a, b[4];
    f32x4* ap = (f32x4*)&a;
    ap[0] = *(const f32x4*)(arow + k0 + 16 * g);
    ap[1] = *(const f32x4*)(arow + k0 + 16 * g + 8);
#pragma unroll
    for (int nt = 0; nt < 4; ++nt) {
      const _Float16* br = lbuf + (nt * 16 + rr) * 32 + 16 * g;
      f32x4* bpv = (f32x4*)&b[nt];
      bpv[0] = *(const f32x4*)br;               // ds_load_b128 x2
      bpv[1] = *(const f32x4*)(br + 8);
    }
#pragma unroll
    for (int nt = 0; nt < 4; ++nt)
      acc[nt] = __builtin_amdgcn_wmma_f32_16x16x32_f16(
          false, a, false, b[nt], (short)0, acc[nt], false, false);
    __syncthreads();                            // done reading buf[i&1]
  }

#pragma unroll
  for (int nt = 0; nt < 4; ++nt) {
    const int n = n0 + nt * 16 + rr;
    const float bb = bias ? bias[n] : 0.f;
#pragma unroll
    for (int e = 0; e < 8; ++e)
      C[(size_t)(m0 + e + 8 * g) * N + n] = acc[nt][e] + bb;
  }
}

// ---------------- attention ----------------
__global__ __launch_bounds__(256) void k_attn_score(
    const float* __restrict__ proj, const float* __restrict__ enc_feat,
    const float* __restrict__ cov, const float* __restrict__ Wc,
    const float* __restrict__ Va, const float* __restrict__ bvp,
    float* __restrict__ score) {
  const int wv = (int)((blockIdx.x * blockDim.x + threadIdx.x) >> 5);
  const int lane = threadIdx.x & 31;
  if (wv >= Bsz * Lsz) return;
  const int b = wv / Lsz;
  const float cv = cov[wv];
  const float* pf = proj + (size_t)b * Ad;
  const float* ef = enc_feat + (size_t)wv * Ad;
  float s = 0.f;
  for (int a = lane; a < Ad; a += 32)
    s += Va[a] * tanhf(pf[a] + ef[a] + cv * Wc[a]);
  s = wredSum(s);
  if (lane == 0) score[wv] = s + bvp[0];
}

__global__ __launch_bounds__(512) void k_attn_softmax(
    const float* __restrict__ score, const float* __restrict__ mask,
    const float* __restrict__ cov_in, float* __restrict__ attn_out,
    float* __restrict__ cov_out) {
  __shared__ float sh[32];
  const int b = blockIdx.x, t = threadIdx.x;
  const float v = (t < Lsz) ? score[b * Lsz + t] : -INFINITY;
  const float m = bredMax(v, sh, 16);
  const float e = (t < Lsz) ? __expf(v - m) : 0.f;
  const float sum = bredSum(e, sh, 16);
  const float a = (t < Lsz) ? (e / sum) * mask[b * Lsz + t] : 0.f;
  const float asum = bredSum(a, sh, 16);
  if (t < Lsz) {
    const float att = a / asum;
    attn_out[b * Lsz + t] = att;
    cov_out[b * Lsz + t] = cov_in[b * Lsz + t] + att;
  }
}

__global__ __launch_bounds__(512) void k_attn_context(
    const float* __restrict__ attn, const float* __restrict__ enc_output,
    float* __restrict__ ctx) {
  __shared__ float sa[Lsz];
  const int b = blockIdx.x, h = threadIdx.x;
  for (int l = threadIdx.x; l < Lsz; l += blockDim.x) sa[l] = attn[b * Lsz + l];
  __syncthreads();
  float s = 0.f;
  const float* eb = enc_output + ((size_t)b * Lsz) * Hd + h;
  for (int l = 0; l < Lsz; ++l) s += sa[l] * eb[(size_t)l * Hd];
  ctx[b * Hd + h] = s;
}

// ---------------- decoder step pieces ----------------
__global__ __launch_bounds__(Dd) void k_build_x(
    const int* __restrict__ dec_inp, int t, const float* __restrict__ embedding,
    const float* __restrict__ ctx, float* __restrict__ x,
    _Float16* __restrict__ x16) {
  const int b = blockIdx.x, i = threadIdx.x;
  float v;
  if (i < Ed) v = embedding[(size_t)dec_inp[b * (Tsteps + 1) + t] * Ed + i];
  else        v = ctx[b * Hd + (i - Ed)];
  x[b * Dd + i] = v;
  x16[b * Dd + i] = (_Float16)v;
}

__global__ __launch_bounds__(256) void k_gru_fuse(
    const float* __restrict__ gx, const float* __restrict__ gu,
    float* __restrict__ h) {
  const int idx = blockIdx.x * 256 + threadIdx.x;
  if (idx >= Bsz * Hd) return;
  const int b = idx / Hd, j = idx % Hd;
  const float* gxb = gx + (size_t)b * 3 * Hd;
  const float* gub = gu + (size_t)b * 3 * Hd;
  const float z  = 1.f / (1.f + __expf(-(gxb[j] + gub[j])));
  const float r  = 1.f / (1.f + __expf(-(gxb[Hd + j] + gub[Hd + j])));
  const float hh = tanhf(gxb[2 * Hd + j] + r * gub[2 * Hd + j]);
  h[idx] = z * h[idx] + (1.f - z) * hh;
}

__global__ __launch_bounds__(1024) void k_pred_softmax(
    const float* __restrict__ logits, float* __restrict__ pred) {
  __shared__ float sh[32];
  const int b = blockIdx.x;
  const float* lb = logits + (size_t)b * Vd;
  float m = -INFINITY;
  for (int v = threadIdx.x; v < Vd; v += 1024) m = fmaxf(m, lb[v]);
  m = bredMax(m, sh, 32);
  float s = 0.f;
  for (int v = threadIdx.x; v < Vd; v += 1024) s += __expf(lb[v] - m);
  s = bredSum(s, sh, 32);
  const float inv = 1.f / s;
  float* pb = pred + (size_t)b * Vd;
  for (int v = threadIdx.x; v < Vd; v += 1024) pb[v] = __expf(lb[v] - m) * inv;
}

__global__ __launch_bounds__(32) void k_pgen(
    const float* __restrict__ ctxn, const float* __restrict__ h,
    const float* __restrict__ x, const float* __restrict__ Wp,
    const float* __restrict__ bp, float* __restrict__ pgen_buf,
    float* __restrict__ out_pgens, int t) {
  const int b = blockIdx.x, lane = threadIdx.x;
  float s = 0.f;
  for (int i = lane; i < Hd; i += 32) s += ctxn[b * Hd + i] * Wp[i];
  for (int i = lane; i < Hd; i += 32) s += h[b * Hd + i] * Wp[Hd + i];
  for (int i = lane; i < Dd; i += 32) s += x[b * Dd + i] * Wp[2 * Hd + i];
  s = wredSum(s);
  if (lane == 0) {
    const float p = 1.f / (1.f + __expf(-(s + bp[0])));
    pgen_buf[b] = p;
    out_pgens[t * Bsz + b] = p;
  }
}

__global__ __launch_bounds__(256) void k_final_vocab(
    const float* __restrict__ pred, const float* __restrict__ pgen,
    float* __restrict__ out_final, int t) {
  const int idx = blockIdx.x * 256 + threadIdx.x;
  if (idx >= Bsz * Vext) return;
  const int b = idx / Vext, v = idx % Vext;
  const float val = (v < Vd) ? pgen[b] * pred[(size_t)b * Vd + v] : 0.f;
  out_final[(size_t)t * Bsz * Vext + idx] = val;
}

__global__ __launch_bounds__(256) void k_final_scatter(
    const int* __restrict__ ext_inp, const float* __restrict__ attn_pre,
    const float* __restrict__ cov_pre, const float* __restrict__ pgen,
    float* __restrict__ out_final, float* __restrict__ out_attns,
    float* __restrict__ out_covs, int t) {
  const int idx = blockIdx.x * 256 + threadIdx.x;
  if (idx >= Bsz * Lsz) return;
  const int b = idx / Lsz;
  const float a = attn_pre[idx];
  out_attns[(size_t)t * Bsz * Lsz + idx] = a;
  out_covs[(size_t)t * Bsz * Lsz + idx] = cov_pre[idx];
  atomicAdd(&out_final[(size_t)t * Bsz * Vext + (size_t)b * Vext + ext_inp[idx]],
            (1.f - pgen[b]) * a);
}

}  // namespace

// ---------------------------------------------------------------------------
extern "C" void kernel_launch(void* const* d_in, const int* in_sizes, int n_in,
                              void* d_out, int out_size, void* d_ws, size_t ws_size,
                              hipStream_t stream) {
  (void)in_sizes; (void)n_in; (void)out_size; (void)ws_size;
  const int*   dec_inp = (const int*)d_in[0];
  const int*   ext_inp = (const int*)d_in[1];
  const float* mask    = (const float*)d_in[2];
  const float* enc_out = (const float*)d_in[4];
  const float* dec_hid = (const float*)d_in[5];
  const float* embed   = (const float*)d_in[6];
  const float* W1 = (const float*)d_in[7];
  const float* b1 = (const float*)d_in[8];
  const float* W2 = (const float*)d_in[9];
  const float* b2 = (const float*)d_in[10];
  const float* Wc = (const float*)d_in[11];
  const float* Va = (const float*)d_in[12];
  const float* bv = (const float*)d_in[13];
  const float* Wg = (const float*)d_in[14];
  const float* Ug = (const float*)d_in[15];
  const float* bg = (const float*)d_in[16];
  const float* Wout = (const float*)d_in[17];
  const float* bout = (const float*)d_in[18];
  const float* Wp = (const float*)d_in[19];
  const float* bp = (const float*)d_in[20];

  // ---- workspace layout ----
  char* ws = (char*)d_ws;
  size_t off = 0;
  auto alloc = [&](size_t bytes) -> char* {
    char* p = ws + off;
    off = (off + bytes + 255) & ~(size_t)255;
    return p;
  };
  _Float16* Wg_t   = (_Float16*)alloc((size_t)3 * Hd * Dd * 2);
  _Float16* Ug_t   = (_Float16*)alloc((size_t)3 * Hd * Hd * 2);
  _Float16* W1_t   = (_Float16*)alloc((size_t)Ad * Hd * 2);
  _Float16* W2_t   = (_Float16*)alloc((size_t)Ad * Hd * 2);
  _Float16* Wout_t = (_Float16*)alloc((size_t)Vd * Hd * 2);
  _Float16* enc16  = (_Float16*)alloc((size_t)Bsz * Lsz * Hd * 2);
  float* enc_feat  = (float*)alloc((size_t)Bsz * Lsz * Ad * 4);
  float* h         = (float*)alloc((size_t)Bsz * Hd * 4);
  _Float16* h16    = (_Float16*)alloc((size_t)Bsz * Hd * 2);
  float* ctx[2]    = {(float*)alloc((size_t)Bsz * Hd * 4),
                      (float*)alloc((size_t)Bsz * Hd * 4)};
  float* x         = (float*)alloc((size_t)Bsz * Dd * 4);
  _Float16* x16    = (_Float16*)alloc((size_t)Bsz * Dd * 2);
  float* attn[2]   = {(float*)alloc((size_t)Bsz * Lsz * 4),
                      (float*)alloc((size_t)Bsz * Lsz * 4)};
  float* cov[2]    = {(float*)alloc((size_t)Bsz * Lsz * 4),
                      (float*)alloc((size_t)Bsz * Lsz * 4)};
  float* score     = (float*)alloc((size_t)Bsz * Lsz * 4);
  float* proj      = (float*)alloc((size_t)Bsz * Ad * 4);
  float* gx        = (float*)alloc((size_t)Bsz * 3 * Hd * 4);
  float* gu        = (float*)alloc((size_t)Bsz * 3 * Hd * 4);
  float* logits    = (float*)alloc((size_t)Bsz * Vd * 4);
  float* pred      = (float*)alloc((size_t)Bsz * Vd * 4);
  float* pgen      = (float*)alloc((size_t)Bsz * 4);

  // ---- output layout (flat, return order) ----
  float* out = (float*)d_out;
  float* out_final = out;
  float* out_attns = out_final + (size_t)Tsteps * Bsz * Vext;
  float* out_covs  = out_attns + (size_t)Tsteps * Bsz * Lsz;
  float* out_hf    = out_covs + (size_t)Tsteps * Bsz * Lsz;
  float* out_ctxf  = out_hf + (size_t)Bsz * Hd;
  float* out_pgens = out_ctxf + (size_t)Bsz * Hd;

  auto cdiv = [](size_t a, size_t b) -> unsigned { return (unsigned)((a + b - 1) / b); };

  // ---- prologue: f16 weight conversion (transposed), enc_feat hoist ----
  k_f32_to_f16_T<<<cdiv((size_t)Dd * 3 * Hd, 256), 256, 0, stream>>>(Wg, Wg_t, Dd, 3 * Hd);
  k_f32_to_f16_T<<<cdiv((size_t)Hd * 3 * Hd, 256), 256, 0, stream>>>(Ug, Ug_t, Hd, 3 * Hd);
  k_f32_to_f16_T<<<cdiv((size_t)Hd * Ad, 256), 256, 0, stream>>>(W1, W1_t, Hd, Ad);
  k_f32_to_f16_T<<<cdiv((size_t)Hd * Ad, 256), 256, 0, stream>>>(W2, W2_t, Hd, Ad);
  k_f32_to_f16_T<<<cdiv((size_t)Hd * Vd, 256), 256, 0, stream>>>(Wout, Wout_t, Hd, Vd);
  k_f32_to_f16<<<cdiv((size_t)Bsz * Lsz * Hd, 256), 256, 0, stream>>>(
      enc_out, enc16, (size_t)Bsz * Lsz * Hd);
  // enc_feat = enc_output @ W2 + b2   (M=12800, N=512, K=512), TDM-staged
  wmma_gemm_tdm<<<dim3(Ad / 64, (Bsz * Lsz) / 32), 64, 0, stream>>>(
      enc16, W2_t, b2, enc_feat, Bsz * Lsz, Ad, Hd);

  k_copy_f32<<<cdiv((size_t)Bsz * Hd, 256), 256, 0, stream>>>(dec_hid, h, (size_t)Bsz * Hd);
  k_zero_f32<<<cdiv((size_t)Bsz * Lsz, 256), 256, 0, stream>>>(cov[0], (size_t)Bsz * Lsz);

  auto attention = [&](const float* hbuf, const float* cov_in, float* ctx_o,
                       float* attn_o, float* cov_o) {
    k_f32_to_f16<<<cdiv((size_t)Bsz * Hd, 256), 256, 0, stream>>>(hbuf, h16, (size_t)Bsz * Hd);
    wmma_gemm_nt<<<dim3(Ad / 64, Bsz / 16), 32, 0, stream>>>(h16, W1_t, b1, proj, Bsz, Ad, Hd);
    k_attn_score<<<cdiv((size_t)Bsz * Lsz * 32, 256), 256, 0, stream>>>(
        proj, enc_feat, cov_in, Wc, Va, bv, score);
    k_attn_softmax<<<Bsz, 512, 0, stream>>>(score, mask, cov_in, attn_o, cov_o);
    k_attn_context<<<Bsz, Hd, 0, stream>>>(attn_o, enc_out, ctx_o);
  };

  // ctx0, attn0, cov0 = attention(dec_hidden, 0)
  attention(h, cov[0], ctx[0], attn[0], cov[0]);

  // ---- scan over T decode steps ----
  int cur = 0;
  for (int t = 0; t < Tsteps; ++t) {
    const int nxt = 1 - cur;
    k_build_x<<<Bsz, Dd, 0, stream>>>(dec_inp, t, embed, ctx[cur], x, x16);
    wmma_gemm_nt<<<dim3(3 * Hd / 64, Bsz / 16), 32, 0, stream>>>(
        x16, Wg_t, bg, gx, Bsz, 3 * Hd, Dd);
    k_f32_to_f16<<<cdiv((size_t)Bsz * Hd, 256), 256, 0, stream>>>(h, h16, (size_t)Bsz * Hd);
    wmma_gemm_nt<<<dim3(3 * Hd / 64, Bsz / 16), 32, 0, stream>>>(
        h16, Ug_t, nullptr, gu, Bsz, 3 * Hd, Hd);
    k_gru_fuse<<<cdiv((size_t)Bsz * Hd, 256), 256, 0, stream>>>(gx, gu, h);  // h := h_new
    // dominant GEMM: h_new @ Wout, TDM-staged through LDS
    k_f32_to_f16<<<cdiv((size_t)Bsz * Hd, 256), 256, 0, stream>>>(h, h16, (size_t)Bsz * Hd);
    wmma_gemm_tdm<<<dim3(Vd / 64, Bsz / 32), 64, 0, stream>>>(
        h16, Wout_t, bout, logits, Bsz, Vd, Hd);
    k_pred_softmax<<<Bsz, 1024, 0, stream>>>(logits, pred);
    attention(h, cov[cur], ctx[nxt], attn[nxt], cov[nxt]);
    k_pgen<<<Bsz, 32, 0, stream>>>(ctx[nxt], h, x, Wp, bp, pgen, out_pgens, t);
    k_final_vocab<<<cdiv((size_t)Bsz * Vext, 256), 256, 0, stream>>>(pred, pgen, out_final, t);
    k_final_scatter<<<cdiv((size_t)Bsz * Lsz, 256), 256, 0, stream>>>(
        ext_inp, attn[cur], cov[cur], pgen, out_final, out_attns, out_covs, t);
    cur = nxt;
  }

  // ---- epilogue: h_f, ctx_f ----
  k_copy_f32<<<cdiv((size_t)Bsz * Hd, 256), 256, 0, stream>>>(h, out_hf, (size_t)Bsz * Hd);
  k_copy_f32<<<cdiv((size_t)Bsz * Hd, 256), 256, 0, stream>>>(ctx[cur], out_ctxf, (size_t)Bsz * Hd);
}
